// selfAttentionV3_64544768524379
// MI455X (gfx1250) — matrix operand order repaired and checked by reading, not compiled, not run on hardware
//
#include <hip/hip_runtime.h>

// ---------------------------------------------------------------------------
// Self-attention forward for MI455X (gfx1250, wave32, WMMA).
//
// X:[4,2048,1024]f32, W:[1024,3072]f32, b:[3072]f32
//   qkv = X@W + b                  (bf16 in d_ws, 48MB)
//   attn = softmax(q@k^T / 32)  -> d_out f32 (mandatory output, 64MB)
//   out  = attn @ v             -> d_out f32
//
// All GEMMs: v_wmma_f32_16x16x32_bf16, fp32 accumulate.
// Block tile 128x128x32, 256 threads = 8 wave32, wave tile 32x64 (2x4 WMMAs).
// Double-buffered LDS. The bf16->bf16 GEMM (q@k^T) uses gfx1250 async
// global->LDS copies (global_load_async_to_lds_b128 + s_wait_asynccnt via
// inline asm, per CDNA5 ISA & CDNA5 ISA); fp32 operands use
// register-staged copies with on-the-fly bf16 conversion.
// ---------------------------------------------------------------------------

typedef __attribute__((ext_vector_type(16))) __bf16          v16bf;
typedef __attribute__((ext_vector_type(8)))  float           v8f;
typedef __attribute__((ext_vector_type(8)))  unsigned short  u16x8;
typedef __attribute__((ext_vector_type(16))) unsigned short  u16x16;

#define BM 128
#define BN 128
#define BK 32
#define THREADS 256

__device__ __forceinline__ unsigned short f32_to_bf16_rne(float f) {
    union { float f; unsigned int u; } c; c.f = f;
    unsigned int u = c.u;
    u += 0x7fffu + ((u >> 16) & 1u);   // round-to-nearest-even
    return (unsigned short)(u >> 16);
}

__device__ __forceinline__ ushort4 load4_bf16(const float* p) {
    float4 f = *(const float4*)p;
    return make_ushort4(f32_to_bf16_rne(f.x), f32_to_bf16_rne(f.y),
                        f32_to_bf16_rne(f.z), f32_to_bf16_rne(f.w));
}
__device__ __forceinline__ ushort4 load4_bf16(const unsigned short* p) {
    return *(const ushort4*)p;
}

__device__ __forceinline__ void store_elem(float* p, float v)          { *p = v; }
__device__ __forceinline__ void store_elem(unsigned short* p, float v) { *p = f32_to_bf16_rne(v); }

// Register staging: issue all global loads first (4 loads in flight), convert,
// then commit to LDS after the WMMAs of the previous tile.
struct Stage4 { ushort4 e[4]; };

// src row-major [128][ld] -> stage (tile 128x32)
template <typename T>
__device__ __forceinline__ void fetch_rows(Stage4& st, const T* src, int ld, int tid) {
#pragma unroll
    for (int i = 0; i < 4; ++i) {
        int v = tid + i * THREADS;
        int r = v >> 3;
        int c = (v & 7) << 2;
        st.e[i] = load4_bf16(src + (long long)r * ld + c);
    }
}
__device__ __forceinline__ void commit_rows(unsigned short* dst, const Stage4& st, int tid) {
#pragma unroll
    for (int i = 0; i < 4; ++i) {
        int v = tid + i * THREADS;
        int r = v >> 3;
        int c = (v & 7) << 2;
        *(ushort4*)(dst + r * BK + c) = st.e[i];
    }
}
// src [K][N] row-major, tile 32x128 -> transposed commit dst[n][k]
template <typename T>
__device__ __forceinline__ void fetch_kn(Stage4& st, const T* src, int ld, int tid) {
#pragma unroll
    for (int i = 0; i < 4; ++i) {
        int v = tid + i * THREADS;
        int k = v >> 5;
        int n = (v & 31) << 2;
        st.e[i] = load4_bf16(src + (long long)k * ld + n);
    }
}
__device__ __forceinline__ void commit_kn_trans(unsigned short* dst, const Stage4& st, int tid) {
#pragma unroll
    for (int i = 0; i < 4; ++i) {
        int v = tid + i * THREADS;
        int k = v >> 5;
        int n = (v & 31) << 2;
        dst[(n + 0) * BK + k] = st.e[i].x;
        dst[(n + 1) * BK + k] = st.e[i].y;
        dst[(n + 2) * BK + k] = st.e[i].z;
        dst[(n + 3) * BK + k] = st.e[i].w;
    }
}

// ---- gfx1250 async global->LDS path (inline asm; ASYNCcnt-tracked) --------

// Generic LDS pointer -> 32-bit LDS byte offset (addrspace(3) ptrtoint).
__device__ __forceinline__ unsigned lds_offset_u32(void* p) {
#if defined(__HIP_DEVICE_COMPILE__)
    typedef __attribute__((address_space(3))) void lds_void;
    return (unsigned)(unsigned long long)(lds_void*)p;
#else
    return (unsigned)(unsigned long long)p;
#endif
}

// GLOBAL_LOAD_ASYNC_TO_LDS_B128 (CDNA5 ISA opcode 98):
// VDST = LDS byte address VGPR, VADDR = 64-bit global address, GV mode.
__device__ __forceinline__ void async_b128(unsigned lds_off, unsigned long long gaddr) {
#if defined(__HIP_DEVICE_COMPILE__)
    asm volatile("global_load_async_to_lds_b128 %0, %1, off"
                 :: "v"(lds_off), "v"(gaddr) : "memory");
#endif
}

__device__ __forceinline__ void wait_async0() {
#if defined(__HIP_DEVICE_COMPILE__)
#if __has_builtin(__builtin_amdgcn_s_wait_asynccnt)
    __builtin_amdgcn_s_wait_asynccnt(0);
#else
    asm volatile("s_wait_asynccnt 0x0" ::: "memory");
#endif
#endif
}

// bf16 row-major 128x32 tile -> LDS, 512 x 16B async chunks (2 per thread).
__device__ __forceinline__ void async_tile_rows(unsigned short* dst,
                                                const unsigned short* src,
                                                int ld, int tid) {
#pragma unroll
    for (int i = 0; i < 2; ++i) {
        int v = tid + i * THREADS;
        int r = v >> 2;
        int c = (v & 3) << 3;
        async_b128(lds_offset_u32(dst + r * BK + c),
                   (unsigned long long)(const void*)(src + (long long)r * ld + c));
    }
}
// Overload so dead branches in fp32 instantiations still type-check.
__device__ __forceinline__ void async_tile_rows(unsigned short*, const float*, int, int) {}

// C[M,N] = scale * (A[M,K] x B) + bias, fp32 accumulation via bf16 WMMA.
//   TA/TB/TC: float (converted to/from bf16) or unsigned short (bf16 bits).
//   TRANSB: B source is [K][N] (transpose into LDS); else [N][K] direct.
//   ASYNC:  use ASYNCcnt global->LDS copies (requires bf16 A/B, !TRANSB).
template <typename TA, typename TB, bool TRANSB, typename TC, bool BIAS, bool ASYNC>
__global__ void __launch_bounds__(THREADS)
gemm_bf16_wmma(const TA* __restrict__ A, long long sAb, int lda,
               const TB* __restrict__ Bm, long long sBb, int ldb,
               TC* __restrict__ C, long long sCb, int ldc,
               const float* __restrict__ bias, float scale, int K)
{
    __shared__ __align__(64) unsigned short As[2][BM * BK];   // [m][k]
    __shared__ __align__(64) unsigned short Bs[2][BN * BK];   // [n][k]

    const int tid   = threadIdx.x;
    const int lane  = tid & 31;
    const int wid   = tid >> 5;
    const int waveM = (wid & 3) * 32;    // 4 waves along M
    const int waveN = (wid >> 2) * 64;   // 2 waves along N

    const int blockM = blockIdx.x * BM;
    const int blockN = blockIdx.y * BN;
    const int batch  = blockIdx.z;

    const TA* Ab = A + (long long)batch * sAb + (long long)blockM * lda;
    const TB* Bb = TRANSB
        ? (Bm + (long long)batch * sBb + blockN)                    // [K][N]
        : (Bm + (long long)batch * sBb + (long long)blockN * ldb);  // [N][K]
    TC* Cb = C + (long long)batch * sCb;

    v8f acc[2][4];
#pragma unroll
    for (int i = 0; i < 2; ++i)
#pragma unroll
        for (int j = 0; j < 4; ++j) acc[i][j] = (v8f){};

    // Preload tile 0 into buffer 0.
    if (ASYNC) {
        async_tile_rows(As[0], Ab, lda, tid);
        async_tile_rows(Bs[0], Bb, ldb, tid);
        wait_async0();
    } else {
        Stage4 sA, sB;
        fetch_rows(sA, Ab, lda, tid);
        if (TRANSB) fetch_kn(sB, Bb, ldb, tid);
        else        fetch_rows(sB, Bb, ldb, tid);
        commit_rows(As[0], sA, tid);
        if (TRANSB) commit_kn_trans(Bs[0], sB, tid);
        else        commit_rows(Bs[0], sB, tid);
    }
    __syncthreads();

    // Fragment geometry (k order used by this kernel, same for A and B; not AMD's table order):
    //  A 16-bit 16x32: lane%16 = M, lane/16 selects K 0-15 vs 16-31, K contiguous
    //  B 16-bit 32x16: lane%16 = N, lane/16 selects K 0-15 vs 16-31, K contiguous
    const int rA  = lane & 15;
    const int kA0 = (lane >> 4) << 4;    // 0 or 16
    const int kB0 = (lane >> 4) << 4;    // 0 or 16

    int cur = 0;
    for (int kt = 0; kt < K; kt += BK) {
        const bool hasNext = (kt + BK) < K;

        // Issue next tile's global traffic before this tile's WMMAs.
        Stage4 sA, sB;
        if (hasNext) {
            if (ASYNC) {
                async_tile_rows(As[cur ^ 1], Ab + kt + BK, lda, tid);
                async_tile_rows(Bs[cur ^ 1], Bb + kt + BK, ldb, tid);
            } else {
                fetch_rows(sA, Ab + kt + BK, lda, tid);
                if (TRANSB) fetch_kn(sB, Bb + (long long)(kt + BK) * ldb, ldb, tid);
                else        fetch_rows(sB, Bb + kt + BK, ldb, tid);
            }
            if (kt + 2 * BK < K)   // stream A two tiles ahead (global_prefetch_b8)
                __builtin_prefetch(Ab + kt + 2 * BK + (long long)(tid & 127) * lda, 0, 0);
        }

        const unsigned short* Ac = As[cur];
        const unsigned short* Bc = Bs[cur];

        v16bf a[2], b[4];
#pragma unroll
        for (int i = 0; i < 2; ++i) {
            const unsigned short* pa = Ac + (waveM + i * 16 + rA) * BK;
            u16x8 lo = *(const u16x8*)(pa + kA0);
            u16x8 hi = *(const u16x8*)(pa + kA0 + 8);
            u16x16 t;
#pragma unroll
            for (int q = 0; q < 8; ++q) { t[q] = lo[q]; t[q + 8] = hi[q]; }
            a[i] = __builtin_bit_cast(v16bf, t);
        }
#pragma unroll
        for (int j = 0; j < 4; ++j) {
            const unsigned short* pb = Bc + (waveN + j * 16 + rA) * BK + kB0;
            b[j] = __builtin_bit_cast(v16bf, *(const u16x16*)pb);
        }

#pragma unroll
        for (int i = 0; i < 2; ++i)
#pragma unroll
            for (int j = 0; j < 4; ++j)
                acc[i][j] = __builtin_amdgcn_wmma_f32_16x16x32_bf16(
                    false, a[i], false, b[j], (short)0, acc[i][j], false, false);

        if (hasNext) {
            if (ASYNC) {
                wait_async0();
            } else {
                commit_rows(As[cur ^ 1], sA, tid);
                if (TRANSB) commit_kn_trans(Bs[cur ^ 1], sB, tid);
                else        commit_rows(Bs[cur ^ 1], sB, tid);
            }
            __syncthreads();
            cur ^= 1;
        }
    }

    // Epilogue. D 16x16 f32 layout: lane%16 = N, lane/16 selects M 0-7 vs 8-15.
    const int mSub = (lane >> 4) << 3;
    const int nSub = lane & 15;
#pragma unroll
    for (int j = 0; j < 4; ++j) {
        const int n = blockN + waveN + j * 16 + nSub;
        float bv = 0.0f;
        if (BIAS) bv = bias[n];
#pragma unroll
        for (int i = 0; i < 2; ++i) {
            const int mBase = blockM + waveM + i * 16 + mSub;
#pragma unroll
            for (int r = 0; r < 8; ++r) {
                float val = acc[i][j][r] * scale + bv;
                store_elem(Cb + (long long)(mBase + r) * ldc + n, val);
            }
        }
    }
}

// In-place row softmax: one 256-thread block (8 wave32) per row of 2048.
__global__ void __launch_bounds__(256)
softmax_rows(float* __restrict__ attn, int ncols)
{
    float* p = attn + (long long)blockIdx.x * ncols;
    const int tid = threadIdx.x, lane = tid & 31, wid = tid >> 5;
    __shared__ float red[8];

    float m = -3.402823466e38f;
    for (int c = tid; c < ncols; c += 256) m = fmaxf(m, p[c]);
#pragma unroll
    for (int o = 16; o > 0; o >>= 1) m = fmaxf(m, __shfl_xor(m, o, 32));
    if (lane == 0) red[wid] = m;
    __syncthreads();
    m = red[0];
#pragma unroll
    for (int w = 1; w < 8; ++w) m = fmaxf(m, red[w]);
    __syncthreads();

    float s = 0.0f;
    for (int c = tid; c < ncols; c += 256) {
        float e = __expf(p[c] - m);
        p[c] = e;
        s += e;
    }
#pragma unroll
    for (int o = 16; o > 0; o >>= 1) s += __shfl_xor(s, o, 32);
    if (lane == 0) red[wid] = s;
    __syncthreads();
    s = red[0];
#pragma unroll
    for (int w = 1; w < 8; ++w) s += red[w];
    const float inv = 1.0f / s;
    for (int c = tid; c < ncols; c += 256) p[c] *= inv;
}

extern "C" void kernel_launch(void* const* d_in, const int* in_sizes, int n_in,
                              void* d_out, int out_size, void* d_ws, size_t ws_size,
                              hipStream_t stream) {
    (void)in_sizes; (void)n_in; (void)out_size; (void)ws_size;

    constexpr int Bn = 4, S = 2048, D = 1024, TD = 3 * D;
    constexpr long long BS = (long long)Bn * S;   // 8192

    const float* X  = (const float*)d_in[0];
    const float* W  = (const float*)d_in[1];
    const float* bb = (const float*)d_in[2];

    float* out  = (float*)d_out;                 // [B,S,D]
    float* attn = out + BS * D;                  // [B,S,S]
    unsigned short* qkv = (unsigned short*)d_ws; // [B*S, 3D] bf16 = 48 MB scratch

    // 1) qkv = X @ W + b  (fp32 in, bf16 out; W is [K][N] -> LDS transpose)
    gemm_bf16_wmma<float, float, true, unsigned short, true, false>
        <<<dim3(BS / BM, TD / BN, 1), THREADS, 0, stream>>>(
            X, 0, D, W, 0, TD, qkv, 0, TD, bb, 1.0f, D);

    // 2) scores = (q @ k^T) / sqrt(D); both operands bf16 -> async LDS copies
    gemm_bf16_wmma<unsigned short, unsigned short, false, float, false, true>
        <<<dim3(S / BM, S / BN, Bn), THREADS, 0, stream>>>(
            qkv, (long long)S * TD, TD,
            qkv + D, (long long)S * TD, TD,
            attn, (long long)S * S, S, nullptr, 0.03125f, D);

    // 3) attn = softmax(scores) in place (attn is a required fp32 output)
    softmax_rows<<<dim3((unsigned)BS), 256, 0, stream>>>(attn, S);

    // 4) out = attn @ v  (attn fp32 -> bf16 in LDS; v is [K][N] -> transpose)
    gemm_bf16_wmma<float, unsigned short, true, float, false, false>
        <<<dim3(S / BM, D / BN, Bn), THREADS, 0, stream>>>(
            attn, (long long)S * S, S,
            qkv + 2 * D, (long long)S * TD, TD,
            out, (long long)S * D, D, nullptr, 1.0f, S);
}
